// SequenceGRU_88888643158715
// MI455X (gfx1250) — matrix operand order repaired and checked by reading, not compiled, not run on hardware
//
#include <hip/hip_runtime.h>
#include <hip/hip_bf16.h>

// ---- problem constants (from reference) ----
#define T_STEPS 2048
#define Bsz     256
#define Fdim    64
#define Hdim    128
#define G3      384      // 3*H
#define MT      16       // batch rows per workgroup
#define NWAVE   8        // waves per workgroup (wave32)
#define KS1     72       // padded K-stride for WiT rows (K=64 + 8 pad)
#define KS2     136      // padded K-stride for Whrz/Whn/h rows (K=128 + 8 pad)

typedef __attribute__((ext_vector_type(16))) __bf16 v16bf;
typedef __attribute__((ext_vector_type(8)))  __bf16 v8bf;
typedef __attribute__((ext_vector_type(8)))  float  v8f;

__device__ __forceinline__ v8f wmma_bf16(v16bf a, v16bf b, v8f c) {
  // D = A(16x32 bf16) * B(32x16 bf16) + C(16x16 f32)
  return __builtin_amdgcn_wmma_f32_16x16x32_bf16(
      /*neg_a=*/false, a, /*neg_b=*/false, b,
      /*c_mod=*/(short)0, c, /*reuse_a=*/false, /*reuse_b=*/false);
}

__device__ __forceinline__ float fast_sigmoid(float x) {
  return __builtin_amdgcn_rcpf(1.0f + __builtin_amdgcn_exp2f(-1.4426950408889634f * x));
}
__device__ __forceinline__ float fast_tanh(float x) {
  float e = __builtin_amdgcn_exp2f(2.8853900817779268f * x);  // exp(2x)
  return 1.0f - 2.0f * __builtin_amdgcn_rcpf(e + 1.0f);
}

// CDNA5 split workgroup barrier: signal (after LDS stores are complete),
// do independent work, then wait.  (ISA: S_BARRIER_SIGNAL/-WAIT, barrier -1)
__device__ __forceinline__ void wg_barrier_signal() {
  asm volatile("s_wait_dscnt 0x0\n\ts_barrier_signal -1" ::: "memory");
}
__device__ __forceinline__ void wg_barrier_wait() {
  asm volatile("s_barrier_wait -1" ::: "memory");
}

// WGP-scope prefetch: default (omitted) scope = WGP -> fills every cache
// level on a miss, unlike __builtin_prefetch which only reaches SE/SYS scope
// (GL2 only).  Non-speculative (address always valid: clamped to t < T).
__device__ __forceinline__ void prefetch_wgp(const void* p) {
  asm volatile("global_prefetch_b8 %0, off" :: "v"(p));
}

// A-fragment (16x32 bf16, M=row, K=32 chunk c) from LDS, row-major bf16 with K-stride ks.
// Layout (ISA 7.12.2): lanes 0-15 row=l hold K = {8h..8h+7} then {16+8h..}, h=half.
__device__ __forceinline__ v16bf ld_afrag(const __bf16* base, int l, int half, int c, int ks) {
  const __bf16* q = base + l * ks + c * 32 + half * 8;
  union { v16bf v; v8bf h2[2]; } u;
  u.h2[0] = *(const v8bf*)q;
  u.h2[1] = *(const v8bf*)(q + 16);
  return u.v;
}

// B-fragment (32x16 bf16, lane = N column, K chunk c) from transposed weights in LDS
// (row n of storage = column n of W, K contiguous). lanes hold K = {8h..8h+7} then {16+8h..}, h=half.
__device__ __forceinline__ v16bf ld_bfrag(const __bf16* base, int l, int half, int c, int ks) {
  const __bf16* q = base + l * ks + c * 32 + half * 8;
  union { v16bf v; v8bf h2[2]; } u;
  u.h2[0] = *(const v8bf*)q;
  u.h2[1] = *(const v8bf*)(q + 16);
  return u.v;
}

// A-fragment for x_t: load f32 from global (row-major, F=64 contiguous), convert to bf16.
__device__ __forceinline__ v16bf ld_xfrag(const float* rowp, int half, int c) {
  const float* q1 = rowp + c * 32 + half * 8;
  const float* q2 = q1 + 16;
  float4 a0 = *(const float4*)q1;
  float4 a1 = *(const float4*)(q1 + 4);
  float4 b0 = *(const float4*)q2;
  float4 b1 = *(const float4*)(q2 + 4);
  v16bf r;
  r[0]  = (__bf16)a0.x; r[1]  = (__bf16)a0.y; r[2]  = (__bf16)a0.z; r[3]  = (__bf16)a0.w;
  r[4]  = (__bf16)a1.x; r[5]  = (__bf16)a1.y; r[6]  = (__bf16)a1.z; r[7]  = (__bf16)a1.w;
  r[8]  = (__bf16)b0.x; r[9]  = (__bf16)b0.y; r[10] = (__bf16)b0.z; r[11] = (__bf16)b0.w;
  r[12] = (__bf16)b1.x; r[13] = (__bf16)b1.y; r[14] = (__bf16)b1.z; r[15] = (__bf16)b1.w;
  return r;
}

__global__ void __launch_bounds__(MT * NWAVE * 2, 1)
gru_seq_kernel(const float* __restrict__ x,      // [T,B,F]
               const float* __restrict__ Wi,     // [F,3H]
               const float* __restrict__ bi,     // [3H]
               const float* __restrict__ Whrz,   // [H,2H]
               const float* __restrict__ Whn,    // [H,H]
               const float* __restrict__ bn,     // [H]
               float* __restrict__ out)          // [T,B,H]
{
  extern __shared__ char smem[];
  __bf16* sWiT   = (__bf16*)smem;                 // [3H][KS1]  (WiT[n][k] = Wi[k][n])
  __bf16* sWhrzT = sWiT   + G3 * KS1;             // [2H][KS2]
  __bf16* sWhnT  = sWhrzT + 2 * Hdim * KS2;       // [H][KS2]
  __bf16* sh     = sWhnT  + Hdim * KS2;           // 2 x [MT][KS2] ping-pong h (bf16)

  const int tid = threadIdx.x;
  const int b0  = blockIdx.x * MT;

  // ---- one-time: stage transposed bf16 weights into LDS ----
  for (int i = tid; i < G3 * Fdim; i += MT * NWAVE * 2) {
    int n = i / Fdim, k = i - n * Fdim;
    sWiT[n * KS1 + k] = (__bf16)Wi[(size_t)k * G3 + n];
  }
  for (int i = tid; i < 2 * Hdim * Hdim; i += MT * NWAVE * 2) {
    int n = i >> 7, k = i & 127;
    sWhrzT[n * KS2 + k] = (__bf16)Whrz[(size_t)k * (2 * Hdim) + n];
  }
  for (int i = tid; i < Hdim * Hdim; i += MT * NWAVE * 2) {
    int n = i >> 7, k = i & 127;
    sWhnT[n * KS2 + k] = (__bf16)Whn[(size_t)k * Hdim + n];
  }
  for (int i = tid; i < 2 * MT * KS2; i += MT * NWAVE * 2) sh[i] = (__bf16)0.0f;  // h0 = 0
  __syncthreads();

  const int w    = tid >> 5;         // wave id: owns H columns [16w, 16w+16)
  const int lane = tid & 31;
  const int l    = lane & 15;
  const int half = lane >> 4;
  const int n0   = w * 16;

  // per-lane gate biases (N = n0 + l column, same for all 8 C/D rows)
  const float brc  = bi[n0 + l];
  const float bzc  = bi[Hdim + n0 + l];
  const float bnc  = bi[2 * Hdim + n0 + l];
  const float bnnc = bn[n0 + l];

  // ---- hoist all time-invariant B fragments into VGPRs (18 x 8 = 144 VGPRs) ----
  v16bf Bxr[2], Bxz[2], Bxn[2];
#pragma unroll
  for (int c = 0; c < 2; ++c) {
    Bxr[c] = ld_bfrag(sWiT + (0 * Hdim + n0) * KS1, l, half, c, KS1);
    Bxz[c] = ld_bfrag(sWiT + (1 * Hdim + n0) * KS1, l, half, c, KS1);
    Bxn[c] = ld_bfrag(sWiT + (2 * Hdim + n0) * KS1, l, half, c, KS1);
  }
  v16bf Bhr[4], Bhz[4], Bhn[4];
#pragma unroll
  for (int c = 0; c < 4; ++c) {
    Bhr[c] = ld_bfrag(sWhrzT + (n0)        * KS2, l, half, c, KS2);
    Bhz[c] = ld_bfrag(sWhrzT + (Hdim + n0) * KS2, l, half, c, KS2);
    Bhn[c] = ld_bfrag(sWhnT  + (n0)        * KS2, l, half, c, KS2);
  }

  v8f hreg = {};   // this wave's f32 h slice, C/D layout: VGPR r -> row r+8*half, col n0+l
  int cur = 0;

  for (int t = 0; t < T_STEPS; ++t) {
    // ---- prefetch x tile 2 steps ahead into WGP$ (global_prefetch_b8, WGP scope).
    // 32 lanes x 1 byte cover the 16 rows x 256B tile (2 cachelines per row).
    int tp = t + 2; if (tp > T_STEPS - 1) tp = T_STEPS - 1;
    prefetch_wgp(x + ((size_t)tp * Bsz + b0 + l) * Fdim + half * 32);

    // ---- input projection for this timestep: xg = x_t @ Wi  (K=64, 2 chunks) ----
    const float* xrow = x + ((size_t)t * Bsz + b0 + l) * Fdim;
    v8f axr = {}, axz = {}, axn = {};
#pragma unroll
    for (int c = 0; c < 2; ++c) {
      v16bf ax = ld_xfrag(xrow, half, c);
      axr = wmma_bf16(ax, Bxr[c], axr);
      axz = wmma_bf16(ax, Bxz[c], axz);
      axn = wmma_bf16(ax, Bxn[c], axn);
    }

    // ---- recurrent projections: h @ Whrz, h @ Whn  (K=128, 4 chunks) ----
    const __bf16* hb = sh + cur * MT * KS2;
    v8f ahr = {}, ahz = {}, ahn = {};
#pragma unroll
    for (int c = 0; c < 4; ++c) {
      v16bf ah = ld_afrag(hb, l, half, c, KS2);
      ahr = wmma_bf16(ah, Bhr[c], ahr);
      ahz = wmma_bf16(ah, Bhz[c], ahz);
      ahn = wmma_bf16(ah, Bhn[c], ahn);
    }

    // ---- gates + state update (f32); h' -> LDS (bf16) for next step's A operand ----
    __bf16* hbn = sh + (cur ^ 1) * MT * KS2;
    float hv[8];
#pragma unroll
    for (int r = 0; r < 8; ++r) {
      float rg = fast_sigmoid(axr[r] + brc + ahr[r]);
      float zg = fast_sigmoid(axz[r] + bzc + ahz[r]);
      float ng = fast_tanh(axn[r] + bnc + rg * (ahn[r] + bnnc));
      hv[r] = (1.0f - zg) * ng + zg * hreg[r];
      hreg[r] = hv[r];
      hbn[(r + 8 * half) * KS2 + n0 + l] = (__bf16)hv[r];
    }

    // ---- split barrier: signal once LDS h' is visible, overlap output stores ----
    wg_barrier_signal();

    float* orow = out + ((size_t)t * Bsz + b0) * Hdim;
#pragma unroll
    for (int r = 0; r < 8; ++r) {
      // ys is a 268MB write-once stream: non-temporal, keep L2 for x + weights
      __builtin_nontemporal_store(hv[r], &orow[(size_t)(r + 8 * half) * Hdim + n0 + l]);
    }

    wg_barrier_wait();
    cur ^= 1;
  }
}

extern "C" void kernel_launch(void* const* d_in, const int* in_sizes, int n_in,
                              void* d_out, int out_size, void* d_ws, size_t ws_size,
                              hipStream_t stream) {
  (void)in_sizes; (void)n_in; (void)out_size; (void)d_ws; (void)ws_size;
  const float* x    = (const float*)d_in[0];
  const float* Wi   = (const float*)d_in[1];
  const float* bi   = (const float*)d_in[2];
  const float* Whrz = (const float*)d_in[3];
  const float* Whn  = (const float*)d_in[4];
  const float* bn   = (const float*)d_in[5];
  float* out = (float*)d_out;

  const size_t smem_bytes =
      (size_t)(G3 * KS1 + 2 * Hdim * KS2 + Hdim * KS2 + 2 * MT * KS2) * sizeof(__bf16);

  hipFuncSetAttribute(reinterpret_cast<const void*>(gru_seq_kernel),
                      hipFuncAttributeMaxDynamicSharedMemorySize, (int)smem_bytes);

  dim3 grid(Bsz / MT);           // 16 workgroups, 16 batch rows each
  dim3 block(MT * NWAVE * 2);    // 256 threads = 8 wave32
  gru_seq_kernel<<<grid, block, smem_bytes, stream>>>(x, Wi, bi, Whrz, Whn, bn, out);
}
